// Feat3dNet_5403068858559
// MI455X (gfx1250) — matrix operand order repaired and checked by reading, not compiled, not run on hardware
//
#include <hip/hip_runtime.h>
#include <math.h>

typedef __attribute__((ext_vector_type(16))) _Float16 v16h;
typedef __attribute__((ext_vector_type(8)))  float    v8f;

#define RADIUS2 4.0f
#define INV_RADIUS 0.5f

// ---------------------------------------------------------------------------
// Fragment layouts (ISA 7.12.2, wave32):
//  A (16x32 f16): lane L holds row M=L%16, khalf=L/16. Element j -> K:
//    j<8 : k = j + 8*khalf            (covers 0..15)
//    j>=8: k = 16 + (j-8) + 8*khalf   (covers 16..31)
//  B (32x16 f16): lane L holds col N=L%16, khalf=L/16; element j -> same k as A (table order)
//  C/D (16x16 f32): vgpr r -> row r + 8*(L/16), col L%16
//
// A-fragment storage order for activations (LDS): element (row,k) at
//   ((kstep*64 + row)*2 + khalf)*16 + j  -> per-lane fragment = 16 contiguous
// halfs (32B) => 2x ds_load_b128, no predication.
// ---------------------------------------------------------------------------
__device__ __forceinline__ int aidx(int row, int k) {
  int ks = k >> 5, kk = k & 31;
  int khalf = (kk >> 3) & 1;
  int j = (kk & 7) | ((kk & 16) >> 1);
  return ((ks * 64 + row) * 2 + khalf) * 16 + j;
}

// One fused layer over a 64-row activation block held in LDS A-fragment form:
// Yfrag(64xN) = relu?(Xfrag(64xK) * Wfrag + bias). 8 waves cooperate; the
// tile count (4*N/16) is a multiple of 8 for every layer we instantiate, so
// the tile loop is a uniform, fully unrolled loop (no divergent exec masking).
template <int K, int N, bool RELU>
__device__ __forceinline__ void layer_lds(const _Float16* Xf, const _Float16* Wf,
                                          const float* bias, _Float16* Yf, int tid) {
  constexpr int KS = (K + 31) / 32;
  constexpr int NTN = N / 16;
  constexpr int NT = 4 * NTN;             // 64 rows -> 4 row tiles
  static_assert(NT % 8 == 0, "tile count must fill 8 waves");
  int wave = tid >> 5, lane = tid & 31;
  int lcol = lane & 15, khalf = lane >> 4;
#pragma unroll
  for (int it = 0; it < NT / 8; ++it) {
    int t = it * 8 + wave;
    int tm = t / NTN, tn = t % NTN;
    int row = tm * 16 + lcol;
    v8f acc = {};
#pragma unroll
    for (int ks = 0; ks < KS; ++ks) {
      v16h a = *(const v16h*)(Xf + ((ks * 64 + row) * 2 + khalf) * 16);
      v16h b = *(const v16h*)(Wf + ((ks * NTN + tn) * 32 + lane) * 16);
      acc = __builtin_amdgcn_wmma_f32_16x16x32_f16(false, a, false, b,
                                                   (short)0, acc, false, false);
    }
    int col = tn * 16 + lcol;
    float bv = bias[col];
    int rbase = tm * 16 + 8 * khalf;
#pragma unroll
    for (int r = 0; r < 8; ++r) {
      float v = acc[r] + bv;
      if (RELU) v = fmaxf(v, 0.f);
      Yf[aidx(rbase + r, col)] = (_Float16)v;   // store into next layer's A-frag
    }
  }
}

// Warm GL2 with this block's weight stream (lowered to global_prefetch_b8).
__device__ __forceinline__ void prefetch_weights(const _Float16* Wf, int halfs, int tid) {
  for (int o = tid * 64; o < halfs; o += 256 * 64)
    __builtin_prefetch(Wf + o, 0, 1);
}

// ---------------------------------------------------------------------------
// Kernel 0: fold BN into weights; emit f16 weights PRE-SWIZZLED into
// B-fragment order (zero-padded K to multiple of 32) + f32 bias.
// ---------------------------------------------------------------------------
struct FoldArgs {
  const float* W[9]; const float* b[9]; const float* g[9];
  const float* bt[9]; const float* mn[9]; const float* vr[9];
  _Float16* Wout[9]; float* bout[9];
  int cin[9], cout[9];
};

__global__ void k_fold(FoldArgs A) {
  int l = blockIdx.x;
  int Ci = A.cin[l], Co = A.cout[l];
  int NTN = Co >> 4, KS = (Ci + 31) >> 5;
  int fsize = KS * NTN * 512;
  _Float16* Wf = A.Wout[l];
  for (int i = threadIdx.x; i < fsize; i += blockDim.x) Wf[i] = (_Float16)0.f;
  __syncthreads();
  for (int i = threadIdx.x; i < Ci * Co; i += blockDim.x) {
    int k = i / Co, n = i % Co;
    float sc = A.g[l][n] * rsqrtf(A.vr[l][n] + 1e-3f);
    int ks = k >> 5, khalf = (k >> 3) & 1, j = (k & 7) | ((k & 16) >> 1);
    int lane = (n & 15) + 16 * khalf, tn = n >> 4;
    Wf[((ks * NTN + tn) * 32 + lane) * 16 + j] = (_Float16)(A.W[l][i] * sc);
  }
  for (int n = threadIdx.x; n < Co; n += blockDim.x) {
    float sc = A.g[l][n] * rsqrtf(A.vr[l][n] + 1e-3f);
    A.bout[l][n] = (A.b[l][n] - A.mn[l][n]) * sc + A.bt[l][n];
  }
}

// ---------------------------------------------------------------------------
// Kernel 1: ball query + group + relative coords. One wave32 per keypoint.
// First 64 in-radius indices in ascending order (== reference top-k).
// ---------------------------------------------------------------------------
__global__ void k_ballquery(const float* __restrict__ pc,   // N x 6
                            const float* __restrict__ kp,   // M x 3
                            float* __restrict__ rel,        // M x 64 x 3
                            int N) {
  __shared__ int sidx[64];
  int m = blockIdx.x;
  int lane = threadIdx.x;
  float kx = kp[m * 3 + 0], ky = kp[m * 3 + 1], kz = kp[m * 3 + 2];
  int fill = 0;
  for (int base = 0; base < N; base += 32) {
    if (fill >= 64) break;
    int i = base + lane;
    float px = pc[i * 6 + 0], py = pc[i * 6 + 1], pz = pc[i * 6 + 2];
    float dx = px - kx, dy = py - ky, dz = pz - kz;
    bool in = (dx * dx + dy * dy + dz * dz) < RADIUS2;
    unsigned long long mask = __ballot(in);
    int before = __popcll(mask & ((1ull << lane) - 1ull));
    int slot = fill + before;
    if (in && slot < 64) sidx[slot] = i;
    fill += __popcll(mask);
    if (fill > 64) fill = 64;
  }
  __syncthreads();
  int f0 = (fill > 0) ? sidx[0] : 0;
  __syncthreads();
  for (int s = lane; s < 64; s += 32)
    if (s >= fill) sidx[s] = f0;
  __syncthreads();
  for (int s = lane; s < 64; s += 32) {
    int j = sidx[s];
    float px = pc[j * 6 + 0], py = pc[j * 6 + 1], pz = pc[j * 6 + 2];
    int o = (m * 64 + s) * 3;
    rel[o + 0] = (px - kx) * INV_RADIUS;
    rel[o + 1] = (py - ky) * INV_RADIUS;
    rel[o + 2] = (pz - kz) * INV_RADIUS;
  }
}

// ---------------------------------------------------------------------------
// Kernel 2: fused det MLP per keypoint: 3->64->128->256 (relu) + row-max.
// LDS fragment-order buffers: X0@0(2048) Y0@2048(4096) Y1@6144(8192)
// Y2@14336(16384) = 30720 halfs = 60KB.
// ---------------------------------------------------------------------------
__global__ void k_det_fused(const float* __restrict__ rel,
                            const _Float16* W0, const float* B0,
                            const _Float16* W1, const float* B1,
                            const _Float16* W2, const float* B2,
                            float* __restrict__ det_mid) {     // M x 256
  __shared__ __align__(32) _Float16 S[30720];
  int m = blockIdx.x, tid = threadIdx.x;
  prefetch_weights(W2, 4 * 16 * 512, tid);   // largest layer first into GL2
  for (int i = tid; i < 2048; i += 256) S[i] = (_Float16)0.f;  // zero-pad X0
  __syncthreads();
  if (tid < 192) {
    int row = tid / 3, c = tid % 3;
    S[aidx(row, c)] = (_Float16)rel[m * 192 + tid];
  }
  __syncthreads();
  layer_lds<3, 64, true>(S + 0, W0, B0, S + 2048, tid);
  __syncthreads();
  layer_lds<64, 128, true>(S + 2048, W1, B1, S + 6144, tid);
  __syncthreads();
  layer_lds<128, 256, true>(S + 6144, W2, B2, S + 14336, tid);
  __syncthreads();
  float mx = -1e30f;
  for (int r = 0; r < 64; ++r) mx = fmaxf(mx, (float)S[14336 + aidx(r, tid)]);
  det_mid[m * 256 + tid] = mx;
}

// ---------------------------------------------------------------------------
// Kernel 3: generic WMMA GEMM with f32 global activations (K,N compile-time).
// A loads are two 8-float runs per kstep -> 4x b128. B is pre-swizzled.
// ---------------------------------------------------------------------------
template <int K, int N, int RELU>
__global__ void k_gemm(const float* __restrict__ X, const _Float16* __restrict__ Wf,
                       const float* __restrict__ bias, float* __restrict__ Y, int R) {
  constexpr int KS = K / 32;
  constexpr int NTN = N / 16;
  int wave = threadIdx.x >> 5, lane = threadIdx.x & 31;
  int tile = blockIdx.x * 4 + wave;
  int ntm = R >> 4;
  if (tile >= NTN * ntm) return;
  int tm = tile / NTN, tn = tile % NTN;
  int lcol = lane & 15, khalf = lane >> 4;
  int row = tm * 16 + lcol;
  v8f acc = {};
#pragma unroll
  for (int ks = 0; ks < KS; ++ks) {
    const float* xp = X + row * K + ks * 32 + 8 * khalf;
    float4 x0 = *(const float4*)(xp + 0);    // k offsets 0..3
    float4 x1 = *(const float4*)(xp + 4);    // 4..7
    float4 x2 = *(const float4*)(xp + 16);   // 16..19
    float4 x3 = *(const float4*)(xp + 20);   // 20..23
    v16h a;
    a[0] = (_Float16)x0.x;  a[1] = (_Float16)x0.y;
    a[2] = (_Float16)x0.z;  a[3] = (_Float16)x0.w;
    a[4] = (_Float16)x1.x;  a[5] = (_Float16)x1.y;
    a[6] = (_Float16)x1.z;  a[7] = (_Float16)x1.w;
    a[8] = (_Float16)x2.x;  a[9] = (_Float16)x2.y;
    a[10] = (_Float16)x2.z; a[11] = (_Float16)x2.w;
    a[12] = (_Float16)x3.x; a[13] = (_Float16)x3.y;
    a[14] = (_Float16)x3.z; a[15] = (_Float16)x3.w;
    v16h b = *(const v16h*)(Wf + ((ks * NTN + tn) * 32 + lane) * 16);
    acc = __builtin_amdgcn_wmma_f32_16x16x32_f16(false, a, false, b,
                                                 (short)0, acc, false, false);
  }
  int col = tn * 16 + lcol;
  float bv = bias[col];
  int rbase = tm * 16 + 8 * khalf;
#pragma unroll
  for (int r = 0; r < 8; ++r) {
    float v = acc[r] + bv;
    if (RELU) v = fmaxf(v, 0.f);
    Y[(rbase + r) * N + col] = v;
  }
}

// ---------------------------------------------------------------------------
// Kernel 4: attention (softplus) + orientation. cos(atan2(y,x)) == x/|v|.
// ---------------------------------------------------------------------------
__global__ void k_att_ori(const float* __restrict__ x,     // M x 64
                          const float* Watt, const float* batt,
                          const float* Wori, const float* bori,
                          float* __restrict__ att_out,     // M
                          float* __restrict__ cs, int M) { // M x 2
  int m = blockIdx.x * blockDim.x + threadIdx.x;
  if (m >= M) return;
  float sa = batt[0], ox = bori[0], oy = bori[1];
  for (int c = 0; c < 64; ++c) {
    float v = x[m * 64 + c];
    sa += v * Watt[c];
    ox += v * Wori[c * 2 + 0];
    oy += v * Wori[c * 2 + 1];
  }
  att_out[m] = (sa > 20.f) ? sa : log1pf(expf(sa));
  float rs = rsqrtf(fmaxf(ox * ox + oy * oy, 1e-8f));
  cs[m * 2 + 0] = ox * rs;   // cos(theta)
  cs[m * 2 + 1] = oy * rs;   // sin(theta)
}

// Kernel 5: rotate rel by -theta -> rel2
__global__ void k_rotate(const float* __restrict__ rel, const float* __restrict__ cs,
                         float* __restrict__ rel2, int total) { // total = M*64
  int i = blockIdx.x * blockDim.x + threadIdx.x;
  if (i >= total) return;
  int m = i >> 6;
  float c = cs[m * 2 + 0], s = cs[m * 2 + 1];
  float rx = rel[i * 3 + 0], ry = rel[i * 3 + 1], rz = rel[i * 3 + 2];
  rel2[i * 3 + 0] = c * rx + s * ry;
  rel2[i * 3 + 1] = -s * rx + c * ry;
  rel2[i * 3 + 2] = rz;
}

// ---------------------------------------------------------------------------
// Kernel 6: fused ext MLP: 3->32->64 (relu), rowmax, concat(y,max)->128,
// 128->128 (no act), rowmax.
// LDS: X0@0(2048) Y0@2048(2048) Y1@4096(4096) CAT@8192(8192) Y2@16384(8192).
// ---------------------------------------------------------------------------
__global__ void k_ext_fused(const float* __restrict__ rel2,
                            const _Float16* W0, const float* B0,
                            const _Float16* W1, const float* B1,
                            const _Float16* W2, const float* B2,
                            float* __restrict__ ext_mid) {   // M x 128
  __shared__ __align__(32) _Float16 S[24576];
  __shared__ float smax[64];
  int m = blockIdx.x, tid = threadIdx.x;
  prefetch_weights(W2, 4 * 8 * 512, tid);
  for (int i = tid; i < 2048; i += 256) S[i] = (_Float16)0.f;
  __syncthreads();
  if (tid < 192) {
    int row = tid / 3, c = tid % 3;
    S[aidx(row, c)] = (_Float16)rel2[m * 192 + tid];
  }
  __syncthreads();
  layer_lds<3, 32, true>(S + 0, W0, B0, S + 2048, tid);
  __syncthreads();
  layer_lds<32, 64, true>(S + 2048, W1, B1, S + 4096, tid);
  __syncthreads();
  if (tid < 64) {
    float mx = -1e30f;
    for (int r = 0; r < 64; ++r) mx = fmaxf(mx, (float)S[4096 + aidx(r, tid)]);
    smax[tid] = mx;
  }
  __syncthreads();
  for (int i = tid; i < 64 * 128; i += 256) {      // concat [y | bcast(max)]
    int row = i >> 7, c = i & 127;
    _Float16 v = (c < 64) ? S[4096 + aidx(row, c)] : (_Float16)smax[c - 64];
    S[8192 + aidx(row, c)] = v;
  }
  __syncthreads();
  layer_lds<128, 128, false>(S + 8192, W2, B2, S + 16384, tid);
  __syncthreads();
  if (tid < 128) {
    float mx = -1e30f;
    for (int r = 0; r < 64; ++r) mx = fmaxf(mx, (float)S[16384 + aidx(r, tid)]);
    ext_mid[m * 128 + tid] = mx;
  }
}

// Kernel 7: copy keypoints + L2-normalize features into d_out
__global__ void k_finalize(const float* __restrict__ kp, const float* __restrict__ feat,
                           float* __restrict__ out, int M) {
  int m = blockIdx.x * blockDim.x + threadIdx.x;
  if (m >= M) return;
  out[m * 3 + 0] = kp[m * 3 + 0];
  out[m * 3 + 1] = kp[m * 3 + 1];
  out[m * 3 + 2] = kp[m * 3 + 2];
  float s = 0.f;
  for (int c = 0; c < 32; ++c) { float v = feat[m * 32 + c]; s += v * v; }
  float rs = rsqrtf(fmaxf(s, 1e-8f));
  for (int c = 0; c < 32; ++c)
    out[M * 3 + m * 32 + c] = feat[m * 32 + c] * rs;
}

// ---------------------------------------------------------------------------
// Host launch. d_in order (pytree insertion order): pointcloud, keypoints,
//   det[0..4]{W,b,gamma,beta,mean,var}, att{W,b}, ori{W,b},
//   ext[0..3]{W,b,gamma,beta,mean,var}
// d_out: [keypoints M*3 | feat M*32 | att M]
// ---------------------------------------------------------------------------
extern "C" void kernel_launch(void* const* d_in, const int* in_sizes, int n_in,
                              void* d_out, int out_size, void* d_ws, size_t ws_size,
                              hipStream_t stream) {
  (void)n_in; (void)out_size; (void)ws_size;
  const float* pc = (const float*)d_in[0];
  const float* kp = (const float*)d_in[1];
  int N = in_sizes[0] / 6;
  int M = in_sizes[1] / 3;

  const int cin[9]  = {3, 64, 128, 256, 128,  3, 32, 128, 128};
  const int cout[9] = {64, 128, 256, 128, 64, 32, 64, 128, 32};
  int pidx[9];
  for (int i = 0; i < 5; ++i) pidx[i] = 2 + 6 * i;        // det layers
  for (int i = 0; i < 4; ++i) pidx[5 + i] = 36 + 6 * i;   // ext layers
  auto P = [&](int i) { return (const float*)d_in[i]; };

  char* ws = (char*)d_ws;
  size_t cur = 0;
  auto alloc = [&](size_t bytes) {
    size_t o = cur; cur += (bytes + 255) & ~(size_t)255; return o;
  };
  size_t woff[9], boff[9];
  for (int l = 0; l < 9; ++l) {
    int KS = (cin[l] + 31) / 32, NTN = cout[l] / 16;
    woff[l] = alloc(sizeof(_Float16) * (size_t)KS * NTN * 512);
    boff[l] = alloc(sizeof(float) * cout[l]);
  }
  size_t relO  = alloc(sizeof(float) * M * 64 * 3);
  size_t rel2O = alloc(sizeof(float) * M * 64 * 3);
  size_t csO   = alloc(sizeof(float) * M * 2);
  size_t dmidO = alloc(sizeof(float) * M * 256);
  size_t dt1O  = alloc(sizeof(float) * M * 128);
  size_t dt2O  = alloc(sizeof(float) * M * 64);
  size_t emidO = alloc(sizeof(float) * M * 128);
  size_t featO = alloc(sizeof(float) * M * 32);

  FoldArgs fa;
  for (int l = 0; l < 9; ++l) {
    fa.W[l]  = P(pidx[l] + 0); fa.b[l]  = P(pidx[l] + 1);
    fa.g[l]  = P(pidx[l] + 2); fa.bt[l] = P(pidx[l] + 3);
    fa.mn[l] = P(pidx[l] + 4); fa.vr[l] = P(pidx[l] + 5);
    fa.Wout[l] = (_Float16*)(ws + woff[l]);
    fa.bout[l] = (float*)(ws + boff[l]);
    fa.cin[l] = cin[l]; fa.cout[l] = cout[l];
  }
  k_fold<<<9, 256, 0, stream>>>(fa);

  k_ballquery<<<M, 32, 0, stream>>>(pc, kp, (float*)(ws + relO), N);

  k_det_fused<<<M, 256, 0, stream>>>((const float*)(ws + relO),
      (const _Float16*)(ws + woff[0]), (const float*)(ws + boff[0]),
      (const _Float16*)(ws + woff[1]), (const float*)(ws + boff[1]),
      (const _Float16*)(ws + woff[2]), (const float*)(ws + boff[2]),
      (float*)(ws + dmidO));

  { int tiles = (M / 16) * (128 / 16);    // det tail: 256->128 relu
    k_gemm<256, 128, 1><<<(tiles + 3) / 4, 128, 0, stream>>>(
        (const float*)(ws + dmidO), (const _Float16*)(ws + woff[3]),
        (const float*)(ws + boff[3]), (float*)(ws + dt1O), M); }
  { int tiles = (M / 16) * (64 / 16);     // det tail: 128->64 relu
    k_gemm<128, 64, 1><<<(tiles + 3) / 4, 128, 0, stream>>>(
        (const float*)(ws + dt1O), (const _Float16*)(ws + woff[4]),
        (const float*)(ws + boff[4]), (float*)(ws + dt2O), M); }

  float* out = (float*)d_out;
  k_att_ori<<<(M + 255) / 256, 256, 0, stream>>>((const float*)(ws + dt2O),
      P(32), P(33), P(34), P(35), out + M * 3 + M * 32, (float*)(ws + csO), M);

  k_rotate<<<(M * 64 + 255) / 256, 256, 0, stream>>>((const float*)(ws + relO),
      (const float*)(ws + csO), (float*)(ws + rel2O), M * 64);

  k_ext_fused<<<M, 256, 0, stream>>>((const float*)(ws + rel2O),
      (const _Float16*)(ws + woff[5]), (const float*)(ws + boff[5]),
      (const _Float16*)(ws + woff[6]), (const float*)(ws + boff[6]),
      (const _Float16*)(ws + woff[7]), (const float*)(ws + boff[7]),
      (float*)(ws + emidO));

  { int tiles = (M / 16) * (32 / 16);     // ext tail: 128->32 no act
    k_gemm<128, 32, 0><<<(tiles + 3) / 4, 128, 0, stream>>>(
        (const float*)(ws + emidO), (const _Float16*)(ws + woff[8]),
        (const float*)(ws + boff[8]), (float*)(ws + featO), M); }

  k_finalize<<<(M + 255) / 256, 256, 0, stream>>>(kp, (const float*)(ws + featO),
                                                  out, M);
}
